// CrossPath_52115133169704
// MI455X (gfx1250) — compile-verified, matrix operand order checked by reading, not run on hardware
//
#include <hip/hip_runtime.h>
#include <hip/hip_bf16.h>
#include <stdint.h>

typedef __bf16 bf16_t;
typedef __attribute__((ext_vector_type(16))) __bf16 v16bf;
typedef __attribute__((ext_vector_type(8)))  __bf16 v8bf;
typedef __attribute__((ext_vector_type(8)))  float  v8f;

#define WMMA_BF16(A, B, Cc) \
    __builtin_amdgcn_wmma_f32_16x16x32_bf16(false, (A), false, (B), (short)0, (Cc), false, false)

static constexpr int Bsz = 4, NT = 2304, C = 256, H = 8, D = 32;
static constexpr int M = Bsz * NT;            // 9216 rows
static constexpr float SCALE = 0.17677669529663689f;  // 32^-0.5

// Fragment load: two contiguous 16B runs per lane (kb = 8*(lane>=16)):
//   elements 0..7  = K = kb .. kb+7 ; elements 8..15 = K = kb+16 .. kb+23
__device__ __forceinline__ v16bf frag_ld(const bf16_t* __restrict__ rowk0, int kb) {
    v8bf lo = *(const v8bf*)(rowk0 + kb);
    v8bf hi = *(const v8bf*)(rowk0 + kb + 16);
    v16bf r;
#pragma unroll
    for (int i = 0; i < 8; ++i) { r[i] = lo[i]; r[i + 8] = hi[i]; }
    return r;
}

__device__ __forceinline__ v16bf frag_ld_lds(const bf16_t* rowk0, int kb) {
    v16bf r;
#pragma unroll
    for (int i = 0; i < 8; ++i) { r[i] = rowk0[kb + i]; r[i + 8] = rowk0[kb + 16 + i]; }
    return r;
}

// ---------------------------------------------------------------- conversion
__global__ void cvt_f32_bf16(const float* __restrict__ src, bf16_t* __restrict__ dst, int n) {
    int i = blockIdx.x * blockDim.x + threadIdx.x;
    if (i < n) dst[i] = (bf16_t)src[i];
}

// ------------------------------------------------------------- fused QKV GEMM
// Y[M,768] = X[M,256] @ Wqkv[768,256]^T ; wave tile 32x64, double-buffered k.
// Scatter into Q,K [B,H,N,D] and V transposed [B,H,D,N].
__global__ __launch_bounds__(256) void qkv_gemm(const bf16_t* __restrict__ X,
                                                const bf16_t* __restrict__ W,
                                                bf16_t* __restrict__ Qb,
                                                bf16_t* __restrict__ Kb,
                                                bf16_t* __restrict__ Vt) {
    int wid  = blockIdx.x * 8 + (threadIdx.x >> 5);
    int lane = threadIdx.x & 31;
    int tN = wid % 12, tM = wid / 12;          // 12 col-supertiles of 64, 288 row tiles of 32
    int m0 = tM * 32, n0 = tN * 64;
    int hi = lane >> 4, lr = lane & 15, kb = hi * 8;

    v8f acc[2][4] = {};
    const bf16_t* arow0 = X + (size_t)(m0 + lr) * C;
    const bf16_t* arow1 = X + (size_t)(m0 + 16 + lr) * C;

    // prologue: stage k0 = 0 fragments
    v16bf a0 = frag_ld(arow0, kb);
    v16bf a1 = frag_ld(arow1, kb);
    v16bf bf[4];
#pragma unroll
    for (int j = 0; j < 4; ++j) bf[j] = frag_ld(W + (size_t)(n0 + j * 16 + lr) * C, kb);

#pragma unroll
    for (int k0 = 0; k0 < C; k0 += 32) {
        // issue next-k loads BEFORE consuming current fragments (double buffer)
        int kn = (k0 + 32 < C) ? k0 + 32 : 0;
        v16bf a0n = frag_ld(arow0 + kn, kb);
        v16bf a1n = frag_ld(arow1 + kn, kb);
        v16bf bn[4];
#pragma unroll
        for (int j = 0; j < 4; ++j) bn[j] = frag_ld(W + (size_t)(n0 + j * 16 + lr) * C + kn, kb);
#pragma unroll
        for (int j = 0; j < 4; ++j) {
            acc[0][j] = WMMA_BF16(a0, bf[j], acc[0][j]);
            acc[1][j] = WMMA_BF16(a1, bf[j], acc[1][j]);
        }
        a0 = a0n; a1 = a1n;
#pragma unroll
        for (int j = 0; j < 4; ++j) bf[j] = bn[j];
    }

#pragma unroll
    for (int i = 0; i < 2; ++i)
#pragma unroll
        for (int j = 0; j < 4; ++j) {
            int c = n0 + j * 16 + lr;          // 0..767
#pragma unroll
            for (int r = 0; r < 8; ++r) {
                int row = m0 + i * 16 + r + 8 * hi;
                int bb = row / NT, t = row % NT;
                bf16_t v = (bf16_t)acc[i][j][r];
                if (c < 256) {
                    int h = c >> 5, d = c & 31;
                    Qb[(((size_t)(bb * H + h)) * NT + t) * D + d] = v;
                } else if (c < 512) {
                    int c2 = c - 256, h = c2 >> 5, d = c2 & 31;
                    Kb[(((size_t)(bb * H + h)) * NT + t) * D + d] = v;
                } else {
                    int c2 = c - 512, h = c2 >> 5, d = c2 & 31;
                    Vt[(((size_t)(bb * H + h)) * D + d) * NT + t] = v;
                }
            }
        }
}

// ------------------------------------------------------------ flash attention
// one wave per (b,h,16-query tile); Q,K row-major [N,D]; V transposed [D,N].
// V-frags for this chunk and K-frags for next chunk are issued before the
// softmax VALU block so global latency overlaps the exp/shfl work.
__global__ __launch_bounds__(32) void attn(const bf16_t* __restrict__ Qb,
                                           const bf16_t* __restrict__ Kb,
                                           const bf16_t* __restrict__ Vt,
                                           const float* __restrict__ beta,
                                           bf16_t* __restrict__ Ob) {
    int id = blockIdx.x;
    int qt = id % (NT / 16), bh = id / (NT / 16);   // bh in [0, 32)
    int h = bh % H, b = bh / H;
    int lane = threadIdx.x & 31;
    int hi = lane >> 4, lr = lane & 15, kb = hi * 8;

    const bf16_t* Qh = Qb + (size_t)bh * NT * D;
    const bf16_t* Kh = Kb + (size_t)bh * NT * D;
    const bf16_t* Vh = Vt + (size_t)bh * D * NT;

    __shared__ bf16_t Pt[16 * 32];

    v16bf qf = frag_ld(Qh + (size_t)(qt * 16 + lr) * D, kb);   // full D=32 window
    float bet = beta[h];

    float mrun[8], lrun[8];
    v8f o0 = {}, o1 = {};
#pragma unroll
    for (int r = 0; r < 8; ++r) { mrun[r] = -1e30f; lrun[r] = 0.0f; }

    // stage first K chunk
    v16bf kf0 = frag_ld(Kh + (size_t)lr * D, kb);
    v16bf kf1 = frag_ld(Kh + (size_t)(16 + lr) * D, kb);

    for (int kc = 0; kc < NT; kc += 32) {
        v8f s0 = {}, s1 = {};
        s0 = WMMA_BF16(qf, kf0, s0);
        s1 = WMMA_BF16(qf, kf1, s1);

        // prefetch: V for this chunk, K for next chunk (overlaps softmax VALU)
        v16bf vf0 = frag_ld(Vh + (size_t)lr * NT + kc, kb);
        v16bf vf1 = frag_ld(Vh + (size_t)(lr + 16) * NT + kc, kb);
        int kcn = (kc + 32 < NT) ? kc + 32 : kc;
        v16bf kf0n = frag_ld(Kh + (size_t)(kcn + lr) * D, kb);
        v16bf kf1n = frag_ld(Kh + (size_t)(kcn + 16 + lr) * D, kb);

        float p0[8], p1[8], cm[8];
#pragma unroll
        for (int r = 0; r < 8; ++r) {
            p0[r] = s0[r] * SCALE + bet;
            p1[r] = s1[r] * SCALE + bet;
            cm[r] = fmaxf(p0[r], p1[r]);
        }
#pragma unroll
        for (int off = 1; off < 16; off <<= 1)
#pragma unroll
            for (int r = 0; r < 8; ++r) cm[r] = fmaxf(cm[r], __shfl_xor(cm[r], off, 32));

        float alpha[8], cs[8];
#pragma unroll
        for (int r = 0; r < 8; ++r) {
            float nm = fmaxf(mrun[r], cm[r]);
            alpha[r] = __expf(mrun[r] - nm);
            mrun[r]  = nm;
            p0[r] = __expf(p0[r] - nm);
            p1[r] = __expf(p1[r] - nm);
            cs[r] = p0[r] + p1[r];
        }
#pragma unroll
        for (int off = 1; off < 16; off <<= 1)
#pragma unroll
            for (int r = 0; r < 8; ++r) cs[r] += __shfl_xor(cs[r], off, 32);
#pragma unroll
        for (int r = 0; r < 8; ++r) {
            lrun[r] = lrun[r] * alpha[r] + cs[r];
            o0[r] *= alpha[r];
            o1[r] *= alpha[r];
        }
        // reshape P (C-layout) -> A-fragment layout via LDS
#pragma unroll
        for (int r = 0; r < 8; ++r) {
            int m = r + 8 * hi;
            Pt[m * 32 + lr]      = (bf16_t)p0[r];
            Pt[m * 32 + lr + 16] = (bf16_t)p1[r];
        }
        __syncthreads();
        v16bf pf = frag_ld_lds(&Pt[lr * 32], kb);
        o0 = WMMA_BF16(pf, vf0, o0);
        o1 = WMMA_BF16(pf, vf1, o1);
        __syncthreads();
        kf0 = kf0n; kf1 = kf1n;
    }
    // finalize: O /= l ; merge heads -> [B,N,C], c = h*32 + d
#pragma unroll
    for (int r = 0; r < 8; ++r) {
        float inv = 1.0f / lrun[r];
        int t = qt * 16 + r + 8 * hi;
        bf16_t* orow = Ob + ((size_t)(b * NT + t)) * C + h * D;
        orow[lr]      = (bf16_t)(o0[r] * inv);
        orow[lr + 16] = (bf16_t)(o1[r] * inv);
    }
}

// ---------------------------------------------- output projection + bias+ReLU
__global__ __launch_bounds__(256) void proj_gemm(const bf16_t* __restrict__ O,
                                                 const bf16_t* __restrict__ W,
                                                 const float* __restrict__ bias,
                                                 float* __restrict__ Z) {
    int wid  = blockIdx.x * 8 + (threadIdx.x >> 5);
    int lane = threadIdx.x & 31;
    int tN = wid % 4, tM = wid / 4;            // 4 col-supertiles of 64, 288 row tiles of 32
    int m0 = tM * 32, n0 = tN * 64;
    int hi = lane >> 4, lr = lane & 15, kb = hi * 8;

    v8f acc[2][4] = {};
    const bf16_t* arow0 = O + (size_t)(m0 + lr) * C;
    const bf16_t* arow1 = O + (size_t)(m0 + 16 + lr) * C;

    v16bf a0 = frag_ld(arow0, kb);
    v16bf a1 = frag_ld(arow1, kb);
    v16bf bf[4];
#pragma unroll
    for (int j = 0; j < 4; ++j) bf[j] = frag_ld(W + (size_t)(n0 + j * 16 + lr) * C, kb);

#pragma unroll
    for (int k0 = 0; k0 < C; k0 += 32) {
        int kn = (k0 + 32 < C) ? k0 + 32 : 0;
        v16bf a0n = frag_ld(arow0 + kn, kb);
        v16bf a1n = frag_ld(arow1 + kn, kb);
        v16bf bn[4];
#pragma unroll
        for (int j = 0; j < 4; ++j) bn[j] = frag_ld(W + (size_t)(n0 + j * 16 + lr) * C + kn, kb);
#pragma unroll
        for (int j = 0; j < 4; ++j) {
            acc[0][j] = WMMA_BF16(a0, bf[j], acc[0][j]);
            acc[1][j] = WMMA_BF16(a1, bf[j], acc[1][j]);
        }
        a0 = a0n; a1 = a1n;
#pragma unroll
        for (int j = 0; j < 4; ++j) bf[j] = bn[j];
    }

#pragma unroll
    for (int i = 0; i < 2; ++i)
#pragma unroll
        for (int j = 0; j < 4; ++j) {
            int c = n0 + j * 16 + lr;
            float bb = bias[c];
#pragma unroll
            for (int r = 0; r < 8; ++r) {
                int row = m0 + i * 16 + r + 8 * hi;
                float v = acc[i][j][r] + bb;
                Z[(size_t)row * C + c] = v > 0.0f ? v : 0.0f;
            }
        }
}

// ---------------------------------------------------------- LayerNorm (inplace)
__global__ __launch_bounds__(32) void layernorm(float* __restrict__ Z,
                                                const float* __restrict__ g,
                                                const float* __restrict__ bq) {
    int row = blockIdx.x, lane = threadIdx.x;
    float* zr = Z + (size_t)row * C;
    float4 a = *(const float4*)(zr + lane * 8);
    float4 c = *(const float4*)(zr + lane * 8 + 4);
    float v[8] = {a.x, a.y, a.z, a.w, c.x, c.y, c.z, c.w};
    float s = 0.f, sq = 0.f;
#pragma unroll
    for (int i = 0; i < 8; ++i) { s += v[i]; sq += v[i] * v[i]; }
#pragma unroll
    for (int off = 1; off < 32; off <<= 1) {
        s  += __shfl_xor(s, off, 32);
        sq += __shfl_xor(sq, off, 32);
    }
    float mu  = s * (1.0f / C);
    float var = sq * (1.0f / C) - mu * mu;
    float rs  = rsqrtf(var + 1e-5f);
#pragma unroll
    for (int i = 0; i < 8; ++i) {
        int cc = lane * 8 + i;
        v[i] = (v[i] - mu) * rs * g[cc] + bq[cc];
    }
    *(float4*)(zr + lane * 8)     = make_float4(v[0], v[1], v[2], v[3]);
    *(float4*)(zr + lane * 8 + 4) = make_float4(v[4], v[5], v[6], v[7]);
}

// ------------------------------- [B,N,C] -> [B,C,96,96] 2x bilinear upsample
__global__ void upsample2x(const float* __restrict__ Y, float* __restrict__ out) {
    int idx = blockIdx.x * 256 + threadIdx.x;
    const int total = Bsz * C * 96 * 96;
    if (idx >= total) return;
    int ox = idx % 96, t = idx / 96;
    int oy = t % 96;  t /= 96;
    int c  = t % C;
    int b  = t / C;
    float sx = ox * 0.5f - 0.25f, sy = oy * 0.5f - 0.25f;
    int x0 = (int)floorf(sx), y0 = (int)floorf(sy);
    float wx = sx - (float)x0, wy = sy - (float)y0;
    int x0c = min(max(x0, 0), 47), x1c = min(max(x0 + 1, 0), 47);
    int y0c = min(max(y0, 0), 47), y1c = min(max(y0 + 1, 0), 47);
    const float* Yb = Y + (size_t)b * NT * C + c;
    float p00 = Yb[(size_t)(y0c * 48 + x0c) * C];
    float p01 = Yb[(size_t)(y0c * 48 + x1c) * C];
    float p10 = Yb[(size_t)(y1c * 48 + x0c) * C];
    float p11 = Yb[(size_t)(y1c * 48 + x1c) * C];
    out[idx] = (1.f - wy) * ((1.f - wx) * p00 + wx * p01) +
               wy * ((1.f - wx) * p10 + wx * p11);
}

// ----------------------------------------------------------------- launcher
extern "C" void kernel_launch(void* const* d_in, const int* in_sizes, int n_in,
                              void* d_out, int out_size, void* d_ws, size_t ws_size,
                              hipStream_t stream) {
    const float* x1    = (const float*)d_in[0];
    const float* x2    = (const float*)d_in[1];
    const float* q1_w  = (const float*)d_in[2];
    const float* kv1_w = (const float*)d_in[3];
    const float* q2_w  = (const float*)d_in[4];
    const float* kv2_w = (const float*)d_in[5];
    const float* beta12 = (const float*)d_in[6];
    const float* beta21 = (const float*)d_in[7];
    const float* p1_w  = (const float*)d_in[8];
    const float* p1_b  = (const float*)d_in[9];
    const float* p2_w  = (const float*)d_in[10];
    const float* p2_b  = (const float*)d_in[11];
    const float* g1    = (const float*)d_in[12];
    const float* b1    = (const float*)d_in[13];
    const float* g2    = (const float*)d_in[14];
    const float* b2    = (const float*)d_in[15];

    char* p = (char*)d_ws;
    auto alloc = [&](size_t bytes) -> void* {
        void* r = (void*)p;
        p += (bytes + 255) & ~(size_t)255;
        return r;
    };
    const size_t NE = (size_t)M * C;                 // 2,359,296 elements
    bf16_t* Xbf[2]  = {(bf16_t*)alloc(NE * 2), (bf16_t*)alloc(NE * 2)};
    bf16_t* Wqkv[2] = {(bf16_t*)alloc(768 * 256 * 2), (bf16_t*)alloc(768 * 256 * 2)};
    bf16_t* Pw[2]   = {(bf16_t*)alloc(256 * 256 * 2), (bf16_t*)alloc(256 * 256 * 2)};
    bf16_t* Qb[2]   = {(bf16_t*)alloc(NE * 2), (bf16_t*)alloc(NE * 2)};
    bf16_t* Kb[2]   = {(bf16_t*)alloc(NE * 2), (bf16_t*)alloc(NE * 2)};
    bf16_t* Vt[2]   = {(bf16_t*)alloc(NE * 2), (bf16_t*)alloc(NE * 2)};
    bf16_t* Ob[2]   = {(bf16_t*)alloc(NE * 2), (bf16_t*)alloc(NE * 2)};
    float*  Z[2]    = {(float*)alloc(NE * 4), (float*)alloc(NE * 4)};
    (void)ws_size; (void)n_in; (void)in_sizes; (void)out_size;

    // f32 -> bf16 conversions
    int nX = (int)NE;
    cvt_f32_bf16<<<(nX + 255) / 256, 256, 0, stream>>>(x1, Xbf[0], nX);
    cvt_f32_bf16<<<(nX + 255) / 256, 256, 0, stream>>>(x2, Xbf[1], nX);
    cvt_f32_bf16<<<(65536 + 255) / 256, 256, 0, stream>>>(q1_w,  Wqkv[0], 65536);
    cvt_f32_bf16<<<(131072 + 255) / 256, 256, 0, stream>>>(kv1_w, Wqkv[0] + 65536, 131072);
    cvt_f32_bf16<<<(65536 + 255) / 256, 256, 0, stream>>>(q2_w,  Wqkv[1], 65536);
    cvt_f32_bf16<<<(131072 + 255) / 256, 256, 0, stream>>>(kv2_w, Wqkv[1] + 65536, 131072);
    cvt_f32_bf16<<<(65536 + 255) / 256, 256, 0, stream>>>(p1_w, Pw[0], 65536);
    cvt_f32_bf16<<<(65536 + 255) / 256, 256, 0, stream>>>(p2_w, Pw[1], 65536);

    // fused QKV projections: (M/32)*(768/64) = 3456 waves / 8 = 432 blocks
    qkv_gemm<<<432, 256, 0, stream>>>(Xbf[0], Wqkv[0], Qb[0], Kb[0], Vt[0]);
    qkv_gemm<<<432, 256, 0, stream>>>(Xbf[1], Wqkv[1], Qb[1], Kb[1], Vt[1]);

    // cross attention: o1 = attn(q1, k2, v2, beta12); o2 = attn(q2, k1, v1, beta21)
    int attnGrid = Bsz * H * (NT / 16);              // 4608
    attn<<<attnGrid, 32, 0, stream>>>(Qb[0], Kb[1], Vt[1], beta12, Ob[0]);
    attn<<<attnGrid, 32, 0, stream>>>(Qb[1], Kb[0], Vt[0], beta21, Ob[1]);

    // output projection + ReLU: (M/32)*(256/64) = 1152 waves / 8 = 144 blocks
    proj_gemm<<<144, 256, 0, stream>>>(Ob[0], Pw[0], p1_b, Z[0]);
    proj_gemm<<<144, 256, 0, stream>>>(Ob[1], Pw[1], p2_b, Z[1]);

    // layernorm in place
    layernorm<<<M, 32, 0, stream>>>(Z[0], g1, b1);
    layernorm<<<M, 32, 0, stream>>>(Z[1], g2, b2);

    // bilinear 2x upsample to [B,C,96,96]; outputs concatenated
    const int totOut = Bsz * C * 96 * 96;            // 9,437,184 per path
    float* out = (float*)d_out;
    upsample2x<<<(totOut + 255) / 256, 256, 0, stream>>>(Z[0], out);
    upsample2x<<<(totOut + 255) / 256, 256, 0, stream>>>(Z[1], out + totOut);
}
